// BiLSTM_CRF_56109452755671
// MI455X (gfx1250) — compile-verified, matrix operand order checked by reading, not run on hardware
//
#include <hip/hip_runtime.h>
#include <hip/hip_bf16.h>

// ---------- problem sizes (fixed by the reference) ----------
#define B_   32
#define S_   512
#define E_   512
#define H_   512
#define G4_  2048   // 4*H
#define T_   16
#define K2_  1024   // E + H (fused [x|h] GEMM K)

#define NBLK_DIR 16         // recurrence blocks per direction
#define HU_BLK   32         // hidden units owned per block (16*32 = 512)
#define W_LDS_BYTES (4 * HU_BLK * K2_ * 2)   // 262144 B weight slab per block

typedef _Float16 f16;
typedef _Float16 v16h __attribute__((ext_vector_type(16)));
typedef _Float16 h8v  __attribute__((ext_vector_type(8)));
typedef float    v8f  __attribute__((ext_vector_type(8)));

// ---------- workspace layout (bytes, all 256B aligned) ----------
#define OFF_XH     0u            // f16 [B,S,E]            16,777,216
#define OFF_W2F    16777216u     // f16 [2048,1024]         4,194,304
#define OFF_W2B    20971520u     // f16 [2048,1024]         4,194,304
#define OFF_WOUT   25165824u     // f16 [16,1024]              32,768
#define OFF_HSF    25198592u     // f16 [B,S,H]            16,777,216
#define OFF_HSB    41975808u     // f16 [B,S,H]            16,777,216
#define OFF_HBUF   58753024u     // f16 [2 dir][2 buf][B,H]   131,072
#define OFF_FEATS  58884096u     // f32 [B,S,T]             1,048,576
#define OFF_BP     59932672u     // i32 [S-1,B,T]           1,046,528
#define OFF_SYNC   60979200u     // u32 counters

// ---------- WMMA helpers ----------
__device__ __forceinline__ v16h load_frag16(const f16* p) {
  // elements 0..7  <-> K = base+koff+0..7   ; elements 8..15 <-> +16..+23
  union { v16h v; h8v h[2]; } u;
  u.h[0] = *(const h8v*)(p);
  u.h[1] = *(const h8v*)(p + 16);
  return u.v;
}

__device__ __forceinline__ v8f wmma16x16x32(v16h a, v16h b, v8f c) {
  return __builtin_amdgcn_wmma_f32_16x16x32_f16(false, a, false, b, (short)0, c,
                                                false, false);
}

__device__ __forceinline__ float sigm(float x) {
  return 1.0f / (1.0f + __expf(-x));
}

// ---------- kernel 0: zero hbuf + sync counters ----------
__global__ void ws_init_k(unsigned int* hbufW, unsigned int* sync) {
  int i = blockIdx.x * blockDim.x + threadIdx.x;
  if (i < (2 * 2 * B_ * H_ * 2) / 4) hbufW[i] = 0u;   // 32768 words
  if (i < 8) sync[i] = 0u;
}

// ---------- kernel 1: embedding gather + f32->f16 ----------
__global__ void embed_gather_k(const int* __restrict__ sent,
                               const float* __restrict__ emb,
                               f16* __restrict__ Xh) {
  const int bs = blockIdx.x;                 // 0 .. B*S-1
  const int e  = threadIdx.x;                // 256 threads, 2 elems each
  const size_t row = (size_t)sent[bs] * E_;
  const size_t o   = (size_t)bs * E_;
  Xh[o + e]       = (f16)emb[row + e];
  Xh[o + e + 256] = (f16)emb[row + e + 256];
}

// ---------- kernel 2: fused weight build  W2[n][k] = k<512 ? Wih : Whh ----------
__global__ void build_w2_k(const float* __restrict__ Wih,
                           const float* __restrict__ Whh,
                           f16* __restrict__ W2) {
  int i = blockIdx.x * blockDim.x + threadIdx.x;
  if (i >= G4_ * K2_) return;
  int n = i >> 10, k = i & 1023;
  float x = (k < E_) ? Wih[(size_t)n * E_ + k] : Whh[(size_t)n * H_ + (k - E_)];
  W2[i] = (f16)x;
}

__global__ void build_wout_k(const float* __restrict__ W, f16* __restrict__ o) {
  int i = blockIdx.x * blockDim.x + threadIdx.x;
  if (i < T_ * K2_) o[i] = (f16)W[i];
}

// ---------- kernel 3: bidirectional LSTM recurrence ----------
// 32 blocks x 128 threads. Blocks 0..15 = forward, 16..31 = backward.
// Each block owns 32 hidden units; its weight slice (4 gates x 32 rows x
// K=1024 f16 = 256KB) is staged once into LDS (CDNA5 320KB/WGP), rows
// interleaved by gate (row = nl*4 + g) so a lane's 4 gate rows sit within
// 0x1800 bytes: every per-step B-fragment is ds_load_b128 from ONE base
// VGPR + 16-bit immediate offset (no per-iteration address VALU).
__global__ void __launch_bounds__(128)
lstm_recurrence_k(const f16* __restrict__ Xh,
                  const f16* __restrict__ W2F, const f16* __restrict__ W2B,
                  const float* __restrict__ bF, const float* __restrict__ bB,
                  f16* __restrict__ hbuf,
                  f16* __restrict__ hsF, f16* __restrict__ hsB,
                  unsigned int* __restrict__ sync) {
  extern __shared__ __align__(16) char smem[];
  f16* Wlds = (f16*)smem;                    // [128 rows][1024 halves]

  const int dir  = blockIdx.x >> 4;
  const int blk  = blockIdx.x & 15;
  const int tid  = threadIdx.x;
  const int wid  = tid >> 5;
  const int lane = tid & 31;
  const int l15  = lane & 15;
  const int koff = (lane >> 4) * 8;          // K-half select per ISA layout
  const int r    = wid & 1;                  // M tile: batches 16r..16r+15
  const int ct   = wid >> 1;                 // N sub-tile (0..1)
  const int h0   = blk * HU_BLK;
  const int hu   = h0 + ct * 16 + l15;       // hidden unit (B column)
  const int brow = r * 16 + l15;             // batch row for A fragment
  const int madd = (lane >= 16) ? 8 : 0;     // D-fragment row offset

  const f16*   W2   = dir ? W2B : W2F;
  const float* bias = dir ? bB : bF;
  f16*         hs   = dir ? hsB : hsF;
  f16*         hb   = hbuf + (size_t)dir * (2 * B_ * H_);
  unsigned int* cnt = sync + dir;

  // ---- stage weight slice into LDS, gate-interleaved: LDS row = nl*4 + g ----
  {
    const int nl_row = tid >> 2;                            // 0..31
    const int g_row  = tid & 3;                             // 0..3
    const uint4* src =
        (const uint4*)(W2 + (size_t)(g_row * H_ + h0 + nl_row) * K2_);
    uint4* dst = (uint4*)(Wlds + (size_t)tid * K2_);        // row tid = nl*4+g
    #pragma unroll 8
    for (int i = 0; i < 128; ++i) dst[i] = src[i];          // 2KB per thread
  }
  __syncthreads();

  // per-lane LDS B-fragment base (time-invariant); gate g at +g*K2_ halves
  const int nl = ct * 16 + l15;
  const f16* wl = Wlds + (size_t)(nl * 4) * K2_ + koff;

  const float bg0 = bias[0 * H_ + hu];
  const float bg1 = bias[1 * H_ + hu];
  const float bg2 = bias[2 * H_ + hu];
  const float bg3 = bias[3 * H_ + hu];

  const f16* xbase = Xh + ((size_t)brow * S_) * E_ + koff;

  v8f c = {};  // cell state, VGPR-resident across all 512 steps

  for (int t = 0; t < S_; ++t) {
    const int s = dir ? (S_ - 1 - t) : t;
    const f16* xrow  = xbase + (size_t)s * E_;
    const f16* hread = hb + (size_t)(t & 1) * (B_ * H_) + (size_t)brow * H_ + koff;
    f16*       hwrt  = hb + (size_t)((t + 1) & 1) * (B_ * H_);

    __builtin_prefetch(xrow, 0, 1);          // global_prefetch_b8

    v8f a0 = {}, a1 = {}, a2 = {}, a3 = {};

    // K chunks 0..15: A from x_t (global, L2-resident)
    #pragma unroll
    for (int kc = 0; kc < 16; ++kc) {
      v16h a = load_frag16(xrow + kc * 32);
      a0 = wmma16x16x32(a, load_frag16(wl + 0 * K2_ + kc * 32), a0);
      a1 = wmma16x16x32(a, load_frag16(wl + 1 * K2_ + kc * 32), a1);
      a2 = wmma16x16x32(a, load_frag16(wl + 2 * K2_ + kc * 32), a2);
      a3 = wmma16x16x32(a, load_frag16(wl + 3 * K2_ + kc * 32), a3);
    }
    // K chunks 16..31: A from h_{t-1} (global exchange buffer)
    #pragma unroll
    for (int kc = 0; kc < 16; ++kc) {
      v16h a = load_frag16(hread + kc * 32);
      a0 = wmma16x16x32(a, load_frag16(wl + 0 * K2_ + (16 + kc) * 32), a0);
      a1 = wmma16x16x32(a, load_frag16(wl + 1 * K2_ + (16 + kc) * 32), a1);
      a2 = wmma16x16x32(a, load_frag16(wl + 2 * K2_ + (16 + kc) * 32), a2);
      a3 = wmma16x16x32(a, load_frag16(wl + 3 * K2_ + (16 + kc) * 32), a3);
    }

    // LSTM cell elementwise: each (lane, v) slot is one (batch, hidden) pair
    #pragma unroll
    for (int v = 0; v < 8; ++v) {
      float ig = sigm (a0[v] + bg0);
      float fg = sigm (a1[v] + bg1);
      float gg = tanhf(a2[v] + bg2);
      float og = sigm (a3[v] + bg3);
      float cv = fg * c[v] + ig * gg;
      c[v] = cv;
      float hv = og * tanhf(cv);
      int  b  = r * 16 + v + madd;
      f16  hh = (f16)hv;
      hwrt[(size_t)b * H_ + hu]          = hh;   // next-step h exchange (L2)
      hs[((size_t)b * S_ + s) * H_ + hu] = hh;   // full h sequence
    }

    // device-scope barrier among the 16 blocks of this direction
    __threadfence();
    __syncthreads();
    if (threadIdx.x == 0) {
      __hip_atomic_fetch_add(cnt, 1u, __ATOMIC_RELEASE, __HIP_MEMORY_SCOPE_AGENT);
      const unsigned int target = (unsigned int)(t + 1) * (unsigned int)NBLK_DIR;
      while (__hip_atomic_load(cnt, __ATOMIC_ACQUIRE, __HIP_MEMORY_SCOPE_AGENT) < target)
        __builtin_amdgcn_s_sleep(1);
    }
    __syncthreads();
  }
}

// ---------- kernel 4: feats = [hf|hb] @ W_out^T + b_out  (M=16384,N=16,K=1024)
__global__ void __launch_bounds__(256)
feats_gemm_k(const f16* __restrict__ hsF, const f16* __restrict__ hsB,
             const f16* __restrict__ WoutH, const float* __restrict__ bout,
             float* __restrict__ feats) {
  const int wid  = threadIdx.x >> 5;
  const int lane = threadIdx.x & 31;
  const int tile = blockIdx.x * 8 + wid;       // 0 .. 1023 (M tiles)
  const int l15  = lane & 15;
  const int koff = (lane >> 4) * 8;
  const int arow = tile * 16 + l15;

  const f16* af = hsF + (size_t)arow * H_ + koff;
  const f16* ab = hsB + (size_t)arow * H_ + koff;
  const f16* wb = WoutH + (size_t)l15 * K2_ + koff;

  v8f acc = {};
  #pragma unroll
  for (int kc = 0; kc < 16; ++kc)
    acc = wmma16x16x32(load_frag16(af + kc * 32), load_frag16(wb + kc * 32), acc);
  #pragma unroll
  for (int kc = 0; kc < 16; ++kc)
    acc = wmma16x16x32(load_frag16(ab + kc * 32),
                       load_frag16(wb + (16 + kc) * 32), acc);

  const int madd = (lane >= 16) ? 8 : 0;
  const float bn = bout[l15];
  #pragma unroll
  for (int v = 0; v < 8; ++v) {
    int row = tile * 16 + v + madd;
    feats[(size_t)row * T_ + l15] = acc[v] + bn;
  }
}

// ---------- kernel 5: Viterbi DP + backtrace (tiny) ----------
__global__ void viterbi_k(const float* __restrict__ feats,
                          const float* __restrict__ trans,
                          const float* __restrict__ startT,
                          const float* __restrict__ stopT,
                          int* __restrict__ bp, float* __restrict__ out) {
  __shared__ float v[B_ * T_];
  __shared__ float vn[B_ * T_];
  __shared__ float tr[T_ * T_];
  __shared__ int lastTag[B_];

  const int tid = threadIdx.x;          // 512 threads
  const int b = tid >> 4, t = tid & 15;
  if (tid < T_ * T_) tr[tid] = trans[tid];
  v[tid] = feats[((size_t)b * S_) * T_ + t] + startT[t];
  __syncthreads();

  for (int s = 1; s < S_; ++s) {
    float best = -3.4e38f; int bi = 0;
    #pragma unroll
    for (int p = 0; p < T_; ++p) {
      float sc = v[b * T_ + p] + tr[p * T_ + t];
      if (sc > best) { best = sc; bi = p; }        // strict > keeps first max
    }
    bp[((size_t)(s - 1) * B_ + b) * T_ + t] = bi;
    vn[tid] = best + feats[((size_t)b * S_ + s) * T_ + t];
    __syncthreads();
    v[tid] = vn[tid];
    __syncthreads();
  }

  if (t == 0) {
    float best = -3.4e38f; int bi = 0;
    for (int p = 0; p < T_; ++p) {
      float sc = v[b * T_ + p] + stopT[p];
      if (sc > best) { best = sc; bi = p; }
    }
    lastTag[b] = bi;
  }
  __syncthreads();

  if (tid < B_) {
    int bb = tid;
    int tag = lastTag[bb];
    out[(size_t)bb * S_ + (S_ - 1)] = (float)tag;
    for (int s = S_ - 1; s >= 1; --s) {
      tag = bp[((size_t)(s - 1) * B_ + bb) * T_ + tag];
      out[(size_t)bb * S_ + (s - 1)] = (float)tag;
    }
  }
}

// ---------- host launcher ----------
extern "C" void kernel_launch(void* const* d_in, const int* in_sizes, int n_in,
                              void* d_out, int out_size, void* d_ws, size_t ws_size,
                              hipStream_t stream) {
  (void)in_sizes; (void)n_in; (void)out_size; (void)ws_size;

  const int*   sent   = (const int*)  d_in[0];
  const float* emb    = (const float*)d_in[1];
  const float* Wih_f  = (const float*)d_in[2];
  const float* Whh_f  = (const float*)d_in[3];
  const float* b_f    = (const float*)d_in[4];
  const float* Wih_b  = (const float*)d_in[5];
  const float* Whh_b  = (const float*)d_in[6];
  const float* b_b    = (const float*)d_in[7];
  const float* W_out  = (const float*)d_in[8];
  const float* b_out  = (const float*)d_in[9];
  const float* transp = (const float*)d_in[10];
  const float* startT = (const float*)d_in[11];
  const float* stopT  = (const float*)d_in[12];

  char* w = (char*)d_ws;
  f16*          Xh    = (f16*)  (w + OFF_XH);
  f16*          W2F   = (f16*)  (w + OFF_W2F);
  f16*          W2B   = (f16*)  (w + OFF_W2B);
  f16*          WoutH = (f16*)  (w + OFF_WOUT);
  f16*          hsF   = (f16*)  (w + OFF_HSF);
  f16*          hsB   = (f16*)  (w + OFF_HSB);
  f16*          hbuf  = (f16*)  (w + OFF_HBUF);
  float*        feats = (float*)(w + OFF_FEATS);
  int*          bp    = (int*)  (w + OFF_BP);
  unsigned int* sync  = (unsigned int*)(w + OFF_SYNC);

  float* outF = (float*)d_out;

  // opt in to 256KB dynamic LDS (CDNA5: 320KB per WGP / workgroup)
  hipFuncSetAttribute(reinterpret_cast<const void*>(lstm_recurrence_k),
                      hipFuncAttributeMaxDynamicSharedMemorySize, W_LDS_BYTES);

  ws_init_k<<<128, 256, 0, stream>>>((unsigned int*)hbuf, sync);
  embed_gather_k<<<B_ * S_, 256, 0, stream>>>(sent, emb, Xh);
  build_w2_k<<<(G4_ * K2_ + 255) / 256, 256, 0, stream>>>(Wih_f, Whh_f, W2F);
  build_w2_k<<<(G4_ * K2_ + 255) / 256, 256, 0, stream>>>(Wih_b, Whh_b, W2B);
  build_wout_k<<<(T_ * K2_ + 255) / 256, 256, 0, stream>>>(W_out, WoutH);
  lstm_recurrence_k<<<32, 128, W_LDS_BYTES, stream>>>(Xh, W2F, W2B, b_f, b_b,
                                                      hbuf, hsF, hsB, sync);
  feats_gemm_k<<<128, 256, 0, stream>>>(hsF, hsB, WoutH, b_out, feats);
  viterbi_k<<<1, 512, 0, stream>>>(feats, transp, startT, stopT, bp, outF);
}
